// LSTMModel_58841051955387
// MI455X (gfx1250) — matrix operand order repaired and checked by reading, not compiled, not run on hardware
//
#include <hip/hip_runtime.h>
#include <stdint.h>

typedef __attribute__((ext_vector_type(16))) __bf16 v16bf;
typedef __attribute__((ext_vector_type(8)))  float  v8f;

#define B_ 2048
#define T_ 256
#define I_ 64
#define H_ 128
#define G_ (4 * H_)

// ---- gfx1250 async global->LDS staging (ASYNCcnt path), with sync fallback ----
#if defined(__has_builtin)
#if __has_builtin(__builtin_amdgcn_global_load_async_to_lds_b128) && \
    __has_builtin(__builtin_amdgcn_s_wait_asynccnt)
#define USE_ASYNC_LDS 1
#endif
#endif

typedef int v4i_ __attribute__((vector_size(16)));
typedef __attribute__((address_space(1))) v4i_* gptr4;
typedef __attribute__((address_space(3))) v4i_* lptr4;

#ifdef USE_ASYNC_LDS
#define STAGE_B(ldsp, gp)                                                     \
  __builtin_amdgcn_global_load_async_to_lds_b128(                             \
      (gptr4)(__attribute__((address_space(1))) void*)(gp),                   \
      (lptr4)(__attribute__((address_space(3))) void*)(ldsp), 0, 0)
#define WAIT_ASYNC() __builtin_amdgcn_s_wait_asynccnt(0)
#else
#define STAGE_B(ldsp, gp) (*(uint4*)(ldsp) = *(const uint4*)(gp))
#define WAIT_ASYNC() ((void)0)
#endif

// ---------- helpers ----------
__device__ __forceinline__ unsigned short f2bf(float f) {
  union { float f; uint32_t u; } x; x.f = f;
  uint32_t u = x.u;
  uint32_t r = u + 0x7FFFu + ((u >> 16) & 1u);   // round-to-nearest-even
  return (unsigned short)(r >> 16);
}
__device__ __forceinline__ float bf2f(unsigned short h) {
  union { uint32_t u; float f; } x; x.u = ((uint32_t)h) << 16;
  return x.f;
}
__device__ __forceinline__ float sigmoidf_(float x) {
  return 1.0f / (1.0f + __expf(-x));
}

union Frag { uint4 q[2]; v16bf v; };

// ---------- generic bf16 WMMA GEMM:  C[M,N] = act(A[M,K] * W[N,K]^T + bias) ----------
// grid.x = M/128, grid.y = N/64, block = 256 (8 waves); wave -> 16(M) x 64(N) tile.
// W tile (64 N-rows x 32 K) is double-buffered in LDS via async global->LDS copies.
// act: 0 none, 1 tanh, 2 relu.  Cf (fp32) and/or Cb (bf16) outputs, either may be null.
__global__ __launch_bounds__(256) void gemm_bf16_wmma(
    const unsigned short* __restrict__ A, int lda,
    const unsigned short* __restrict__ W, int ldw,
    const float* __restrict__ bias,
    float* __restrict__ Cf, unsigned short* __restrict__ Cb, int ldc,
    int K, int act)
{
  __shared__ unsigned short Bs[2][64 * 32];   // 2 x 4 KB double buffer
  const int wave = threadIdx.x >> 5;
  const int lane = threadIdx.x & 31;
  const long m0 = (long)blockIdx.x * 128 + wave * 16;
  const int  n0 = blockIdx.y * 64;

  v8f acc[4] = { v8f{}, v8f{}, v8f{}, v8f{} };

  // A fragment: lane<16 -> K {0..7,16..23}; lane>=16 -> K {8..15,24..31}
  const long aRow = m0 + (lane & 15);
  const int  offA = (lane >> 4) << 3;   // 0 or 8 elements
  // B fragment: lane<16 -> K {0..7,16..23}; lane>=16 -> K {8..15,24..31}
  const int  offB = (lane >> 4) << 3;   // 0 or 8 elements
  const unsigned short* Aptr = A + aRow * (long)lda;

  // cooperative staging: 256 threads x 16B = 64 rows x 64B tile
  const int stg_r = threadIdx.x >> 2;            // 0..63  (N row in tile)
  const int stg_c = (threadIdx.x & 3) << 3;      // 0,8,16,24 (K elements)
  const unsigned short* Wstg = W + (long)(n0 + stg_r) * ldw + stg_c;
  unsigned short* stgDst0 = &Bs[0][stg_r * 32 + stg_c];
  unsigned short* stgDst1 = &Bs[1][stg_r * 32 + stg_c];

  STAGE_B(stgDst0, Wstg);                 // prologue: tile for k0 = 0
  int cur = 0;
  for (int k0 = 0; k0 < K; k0 += 32) {
    WAIT_ASYNC();                         // this wave's pending tile landed in LDS
    __syncthreads();                      // all waves' staging visible
    if (k0 + 32 < K) {
      STAGE_B(cur ? stgDst0 : stgDst1, Wstg + k0 + 32);   // prefetch next tile
      __builtin_prefetch(Aptr + k0 + 32, 0, 1);           // global_prefetch_b8
    }

    Frag a;
    a.q[0] = *(const uint4*)&Aptr[k0 + offA];
    a.q[1] = *(const uint4*)&Aptr[k0 + offA + 16];

    const unsigned short* Bcur = &Bs[cur][0];
    #pragma unroll
    for (int j = 0; j < 4; ++j) {
      const int rowB = j * 16 + (lane & 15);
      Frag b;
      b.q[0] = *(const uint4*)&Bcur[rowB * 32 + offB];
      b.q[1] = *(const uint4*)&Bcur[rowB * 32 + offB + 16];
      acc[j] = __builtin_amdgcn_wmma_f32_16x16x32_bf16(
          false, a.v, false, b.v, (short)0, acc[j], false, false);
    }
    cur ^= 1;
  }

  // C layout: VGPR r -> lanes 0-15: (M=r, N=lane), lanes 16-31: (M=r+8, N=lane-16)
  const int nc    = lane & 15;
  const int rbase = (lane >> 4) << 3;
  #pragma unroll
  for (int j = 0; j < 4; ++j) {
    const int col = n0 + j * 16 + nc;
    const float bv = bias ? bias[col] : 0.0f;
    #pragma unroll
    for (int r = 0; r < 8; ++r) {
      const long row = m0 + rbase + r;
      float v = acc[j][r] + bv;
      if (act == 1) v = tanhf(v);
      else if (act == 2) v = fmaxf(v, 0.0f);
      const long idx = row * (long)ldc + col;
      if (Cf) Cf[idx] = v;
      if (Cb) Cb[idx] = f2bf(v);
    }
  }
}

// ---------- elementwise / small kernels ----------
__global__ void cast_f32_bf16(const float* __restrict__ src,
                              unsigned short* __restrict__ dst, long n) {
  long i = (long)blockIdx.x * blockDim.x + threadIdx.x;
  if (i < n) dst[i] = f2bf(src[i]);
}

__global__ void add_vec(const float* __restrict__ a, const float* __restrict__ b,
                        float* __restrict__ o, int n) {
  int i = blockIdx.x * blockDim.x + threadIdx.x;
  if (i < n) o[i] = a[i] + b[i];
}

__global__ void zero_f32(float* __restrict__ p, long n) {
  long i = (long)blockIdx.x * blockDim.x + threadIdx.x;
  if (i < n) p[i] = 0.0f;
}
__global__ void zero_bf(unsigned short* __restrict__ p, long n) {
  long i = (long)blockIdx.x * blockDim.x + threadIdx.x;
  if (i < n) p[i] = 0;
}

// fused LSTM cell: gates = pre[b*T+t] + g[b]; i,f,g,o order (PyTorch).
__global__ __launch_bounds__(256) void lstm_cell(
    const float* __restrict__ pre, const float* __restrict__ g,
    float* __restrict__ c, unsigned short* __restrict__ h_bf,
    unsigned short* __restrict__ hseq, int t)
{
  const int idx = blockIdx.x * blockDim.x + threadIdx.x;  // B_*H_
  const int b  = idx >> 7;          // / H_
  const int hh = idx & (H_ - 1);
  const long prow = ((long)b * T_ + t) * G_;
  const long grow = (long)b * G_;
  float iv = sigmoidf_(pre[prow + hh]          + g[grow + hh]);
  float fv = sigmoidf_(pre[prow + H_ + hh]     + g[grow + H_ + hh]);
  float gv = tanhf    (pre[prow + 2 * H_ + hh] + g[grow + 2 * H_ + hh]);
  float ov = sigmoidf_(pre[prow + 3 * H_ + hh] + g[grow + 3 * H_ + hh]);
  float cn = fv * c[idx] + iv * gv;
  c[idx] = cn;
  unsigned short hb = f2bf(ov * tanhf(cn));
  h_bf[idx] = hb;
  hseq[((long)b * T_ + t) * H_ + hh] = hb;
}

// out[row] = dot(A[row, 0:K] (bf16), w (f32)) + bias[0];  one wave per row, 8 rows/block
__global__ __launch_bounds__(256) void row_dot_bf(
    const unsigned short* __restrict__ A, int lda,
    const float* __restrict__ w, const float* __restrict__ bias,
    float* __restrict__ out, int K)
{
  const int lane = threadIdx.x & 31;
  const long row = (long)blockIdx.x * 8 + (threadIdx.x >> 5);
  float s = 0.0f;
  for (int k = lane; k < K; k += 32) s += bf2f(A[row * (long)lda + k]) * w[k];
  #pragma unroll
  for (int o = 16; o > 0; o >>= 1) s += __shfl_xor(s, o, 32);
  if (lane == 0) out[row] = s + bias[0];
}

// softmax over T per batch row; block = T_ = 256 threads
__global__ __launch_bounds__(256) void softmax_time(
    const float* __restrict__ sc, float* __restrict__ attn)
{
  __shared__ float sm[256];
  const int b = blockIdx.x, t = threadIdx.x;
  float v = sc[(long)b * T_ + t];
  sm[t] = v; __syncthreads();
  for (int s = 128; s > 0; s >>= 1) { if (t < s) sm[t] = fmaxf(sm[t], sm[t + s]); __syncthreads(); }
  float m = sm[0]; __syncthreads();
  float e = __expf(v - m);
  sm[t] = e; __syncthreads();
  for (int s = 128; s > 0; s >>= 1) { if (t < s) sm[t] += sm[t + s]; __syncthreads(); }
  attn[(long)b * T_ + t] = e / sm[0];
}

// context[b,h] = sum_t hseq[b,t,h] * attn[b,t];  block = H_ threads, grid = B_
__global__ __launch_bounds__(128) void context_kernel(
    const unsigned short* __restrict__ hseq, const float* __restrict__ attn,
    unsigned short* __restrict__ ctx_bf)
{
  const int b = blockIdx.x, h = threadIdx.x;
  const unsigned short* hp = hseq + ((long)b * T_) * H_ + h;
  const float* ap = attn + (long)b * T_;
  float s = 0.0f;
  for (int t = 0; t < T_; ++t) s += bf2f(hp[(long)t * H_]) * ap[t];
  ctx_bf[(long)b * H_ + h] = f2bf(s);
}

// ---------- host ----------
extern "C" void kernel_launch(void* const* d_in, const int* in_sizes, int n_in,
                              void* d_out, int out_size, void* d_ws, size_t ws_size,
                              hipStream_t stream) {
  (void)in_sizes; (void)n_in; (void)out_size; (void)ws_size;
  const float* x     = (const float*)d_in[0];
  const float* w_ih0 = (const float*)d_in[1];
  const float* w_hh0 = (const float*)d_in[2];
  const float* b_ih0 = (const float*)d_in[3];
  const float* b_hh0 = (const float*)d_in[4];
  const float* w_ih1 = (const float*)d_in[5];
  const float* w_hh1 = (const float*)d_in[6];
  const float* b_ih1 = (const float*)d_in[7];
  const float* b_hh1 = (const float*)d_in[8];
  const float* aw1   = (const float*)d_in[9];
  const float* ab1   = (const float*)d_in[10];
  const float* aw2   = (const float*)d_in[11];
  const float* ab2   = (const float*)d_in[12];
  const float* fw1   = (const float*)d_in[13];
  const float* fb1   = (const float*)d_in[14];
  const float* fw2   = (const float*)d_in[15];
  const float* fb2   = (const float*)d_in[16];
  const float* fw3   = (const float*)d_in[17];
  const float* fb3   = (const float*)d_in[18];
  float* out = (float*)d_out;

  char* p = (char*)d_ws;
  auto carve = [&](size_t bytes) -> char* {
    char* r = p; p += (bytes + 255) & ~(size_t)255; return r;
  };
  const long NX = (long)B_ * T_ * I_;          // 33.5M
  const long NS = (long)B_ * T_ * H_;          // 67M  (seq activations)
  const long NP = (long)B_ * T_ * G_;          // 268M (gate pre-activations)

  unsigned short* x_bf    = (unsigned short*)carve(NX * 2);
  unsigned short* wih0_bf = (unsigned short*)carve((size_t)G_ * I_ * 2);
  unsigned short* whh0_bf = (unsigned short*)carve((size_t)G_ * H_ * 2);
  unsigned short* wih1_bf = (unsigned short*)carve((size_t)G_ * H_ * 2);
  unsigned short* whh1_bf = (unsigned short*)carve((size_t)G_ * H_ * 2);
  unsigned short* aw1_bf  = (unsigned short*)carve((size_t)H_ * H_ * 2);
  unsigned short* fw1_bf  = (unsigned short*)carve((size_t)H_ * H_ * 2);
  unsigned short* fw2_bf  = (unsigned short*)carve((size_t)(H_/2) * H_ * 2);
  float* bias0 = (float*)carve(G_ * 4);
  float* bias1 = (float*)carve(G_ * 4);
  float* pre   = (float*)carve(NP * 4);                 // reused layer0 -> layer1
  float* gbuf  = (float*)carve((size_t)B_ * G_ * 4);
  float* cstat = (float*)carve((size_t)B_ * H_ * 4);
  unsigned short* h_bf   = (unsigned short*)carve((size_t)B_ * H_ * 2);
  unsigned short* h0seq  = (unsigned short*)carve(NS * 2);   // reused as s1 later
  unsigned short* h1seq  = (unsigned short*)carve(NS * 2);
  float* scores = (float*)carve((size_t)B_ * T_ * 4);
  float* attn   = (float*)carve((size_t)B_ * T_ * 4);
  unsigned short* ctx_bf = (unsigned short*)carve((size_t)B_ * H_ * 2);
  unsigned short* fc1_bf = (unsigned short*)carve((size_t)B_ * H_ * 2);
  unsigned short* fc2_bf = (unsigned short*)carve((size_t)B_ * (H_/2) * 2);

  auto cast = [&](const float* s, unsigned short* d, long n) {
    cast_f32_bf16<<<dim3((unsigned)((n + 255) / 256)), 256, 0, stream>>>(s, d, n);
  };
  // 1) casts + combined biases
  cast(x, x_bf, NX);
  cast(w_ih0, wih0_bf, (long)G_ * I_);
  cast(w_hh0, whh0_bf, (long)G_ * H_);
  cast(w_ih1, wih1_bf, (long)G_ * H_);
  cast(w_hh1, whh1_bf, (long)G_ * H_);
  cast(aw1, aw1_bf, (long)H_ * H_);
  cast(fw1, fw1_bf, (long)H_ * H_);
  cast(fw2, fw2_bf, (long)(H_/2) * H_);
  add_vec<<<dim3(2), 256, 0, stream>>>(b_ih0, b_hh0, bias0, G_);
  add_vec<<<dim3(2), 256, 0, stream>>>(b_ih1, b_hh1, bias1, G_);

  const long BT = (long)B_ * T_;
  // 2) layer-0 input projection: pre = x * w_ih0^T + bias0   [BT x 512], K=64
  gemm_bf16_wmma<<<dim3((unsigned)(BT / 128), G_ / 64), 256, 0, stream>>>(
      x_bf, I_, wih0_bf, I_, bias0, pre, nullptr, G_, I_, 0);

  // 3) layer-0 recurrence
  zero_bf<<<dim3((B_ * H_) / 256), 256, 0, stream>>>(h_bf, (long)B_ * H_);
  zero_f32<<<dim3((B_ * H_) / 256), 256, 0, stream>>>(cstat, (long)B_ * H_);
  for (int t = 0; t < T_; ++t) {
    gemm_bf16_wmma<<<dim3(B_ / 128, G_ / 64), 256, 0, stream>>>(
        h_bf, H_, whh0_bf, H_, nullptr, gbuf, nullptr, G_, H_, 0);
    lstm_cell<<<dim3((B_ * H_) / 256), 256, 0, stream>>>(pre, gbuf, cstat, h_bf, h0seq, t);
  }

  // 4) layer-1 input projection: pre = h0seq * w_ih1^T + bias1, K=128 (reuses pre)
  gemm_bf16_wmma<<<dim3((unsigned)(BT / 128), G_ / 64), 256, 0, stream>>>(
      h0seq, H_, wih1_bf, H_, bias1, pre, nullptr, G_, H_, 0);

  // 5) layer-1 recurrence
  zero_bf<<<dim3((B_ * H_) / 256), 256, 0, stream>>>(h_bf, (long)B_ * H_);
  zero_f32<<<dim3((B_ * H_) / 256), 256, 0, stream>>>(cstat, (long)B_ * H_);
  for (int t = 0; t < T_; ++t) {
    gemm_bf16_wmma<<<dim3(B_ / 128, G_ / 64), 256, 0, stream>>>(
        h_bf, H_, whh1_bf, H_, nullptr, gbuf, nullptr, G_, H_, 0);
    lstm_cell<<<dim3((B_ * H_) / 256), 256, 0, stream>>>(pre, gbuf, cstat, h_bf, h1seq, t);
  }

  // 6) attention: s1 = tanh(h1seq * aw1^T + ab1)  (bf16, reuses h0seq buffer)
  unsigned short* s1_bf = h0seq;
  gemm_bf16_wmma<<<dim3((unsigned)(BT / 128), H_ / 64), 256, 0, stream>>>(
      h1seq, H_, aw1_bf, H_, ab1, nullptr, s1_bf, H_, H_, 1);
  // scores = s1 * aw2^T + ab2
  row_dot_bf<<<dim3((unsigned)(BT / 8)), 256, 0, stream>>>(s1_bf, H_, aw2, ab2, scores, H_);
  // softmax over T, then context
  softmax_time<<<dim3(B_), 256, 0, stream>>>(scores, attn);
  context_kernel<<<dim3(B_), 128, 0, stream>>>(h1seq, attn, ctx_bf);

  // 7) MLP head
  gemm_bf16_wmma<<<dim3(B_ / 128, H_ / 64), 256, 0, stream>>>(
      ctx_bf, H_, fw1_bf, H_, fb1, nullptr, fc1_bf, H_, H_, 2);
  gemm_bf16_wmma<<<dim3(B_ / 128, (H_/2) / 64), 256, 0, stream>>>(
      fc1_bf, H_, fw2_bf, H_, fb2, nullptr, fc2_bf, H_ / 2, H_, 2);
  row_dot_bf<<<dim3(B_ / 8), 256, 0, stream>>>(fc2_bf, H_ / 2, fw3, fb3, out, H_ / 2);
}
